// Connect4CNN_Mk5_62680752718012
// MI455X (gfx1250) — matrix operand order repaired and checked by reading, not compiled, not run on hardware
//
#include <hip/hip_runtime.h>
#include <hip/hip_bf16.h>
#include <math.h>

typedef __attribute__((ext_vector_type(16))) _Float16 v16h;
typedef __attribute__((ext_vector_type(8)))  _Float16 v8h;
typedef __attribute__((ext_vector_type(8)))  float    v8f;

union AFrag { v16h v; v8h h2[2]; _Float16 e[16]; };
union CFrag { v8f  v; float e[8]; };
union X32   { v8h  h[4]; _Float16 e[32]; };

#define TB     16          // batch samples per block
#define HW     42          // 6*7
#define NM     (TB*HW)     // 672 GEMM rows per block
#define NE     (NM*32)     // 21504 f16 per activation buffer
#define MTILES 42

// f16 weight workspace layout (element offsets), after the f-tile region
#define OFF_W0   0                        // conv0 [o][k], K=tap*3+ci padded to 32
#define OFF_CONV 1024                     // 6 layers x [tap][o][ci]
#define OFF_SD   (OFF_CONV + 6*9216)      // sd_w   [128][1344]
#define OFF_PW1  (OFF_SD + 128*1344)      // p_w1   [128][128]
#define OFF_PW2  (OFF_PW1 + 128*128)      // p_w2   [128][128]
#define OFF_VW1  (OFF_PW2 + 128*128)      // v_w1   [64][128]
#define OFF_VW2  (OFF_VW1 + 64*128)       // v_w2   [64][64]

__device__ __forceinline__ v8f wmma_f16(v16h a, v16h b, v8f c) {
    return __builtin_amdgcn_wmma_f32_16x16x32_f16(false, a, false, b, (short)0, c, false, false);
}

// =====================  Kernel 0: one-shot weight transpose/convert  =====================
__global__ void __launch_bounds__(256)
prep_weights(const float* __restrict__ conv0_w,
             const float* __restrict__ cw0, const float* __restrict__ cw1,
             const float* __restrict__ cw2, const float* __restrict__ cw3,
             const float* __restrict__ cw4, const float* __restrict__ cw5,
             const float* __restrict__ sd_w, const float* __restrict__ p_w1,
             const float* __restrict__ p_w2, const float* __restrict__ v_w1,
             const float* __restrict__ v_w2, _Float16* __restrict__ wsw) {
    const int stride = gridDim.x * blockDim.x;
    const int t0 = blockIdx.x * blockDim.x + threadIdx.x;
    for (int i = t0; i < 1024; i += stride) {
        int o = i >> 5, k = i & 31;
        float v = 0.f;
        if (k < 27) { int tap = k / 3, ci = k - tap * 3; v = conv0_w[(o * 3 + ci) * 9 + tap]; }
        wsw[OFF_W0 + i] = (_Float16)v;
    }
    const float* cws[6] = {cw0, cw1, cw2, cw3, cw4, cw5};
#pragma unroll
    for (int l = 0; l < 6; ++l) {
        const float* w = cws[l];
        for (int i = t0; i < 9216; i += stride) {
            int tap = i >> 10, o = (i >> 5) & 31, ci = i & 31;
            wsw[OFF_CONV + l * 9216 + i] = (_Float16)w[(o * 32 + ci) * 9 + tap];
        }
    }
    for (int i = t0; i < 128 * 1344; i += stride) wsw[OFF_SD  + i] = (_Float16)sd_w[i];
    for (int i = t0; i < 128 * 128;  i += stride) wsw[OFF_PW1 + i] = (_Float16)p_w1[i];
    for (int i = t0; i < 128 * 128;  i += stride) wsw[OFF_PW2 + i] = (_Float16)p_w2[i];
    for (int i = t0; i < 64 * 128;   i += stride) wsw[OFF_VW1 + i] = (_Float16)v_w1[i];
    for (int i = t0; i < 64 * 64;    i += stride) wsw[OFF_VW2 + i] = (_Float16)v_w2[i];
}

// =====================  Kernel 1: fused conv tower  =====================
// act layout: act[m][ci], m = s*42+pos (32 contiguous f16 = 64B/row)
// B fragments load straight from global f16 [tap][o][ci] (L2-resident, hoisted)

__device__ __forceinline__ void conv_layer(const _Float16* __restrict__ in,
                                           _Float16* __restrict__ out,
                                           const _Float16* __restrict__ wsrc,
                                           int lane, int wid) {
    const int koff = (lane & 16) ? 16 : 0;
    const int aoff = (lane & 16) ? 16 : 0;
    v16h Bf[9][2];
#pragma unroll
    for (int tap = 0; tap < 9; ++tap)
#pragma unroll
        for (int nt = 0; nt < 2; ++nt) {
            AFrag b;
            const v8h* p = (const v8h*)(wsrc + tap * 1024 + (nt * 16 + (lane & 15)) * 32 + koff);
            b.h2[0] = p[0]; b.h2[1] = p[1];
            Bf[tap][nt] = b.v;
        }
    const v8h zero = {};
    for (int mt = wid; mt < MTILES; mt += 8) {
        const int m   = mt * 16 + (lane & 15);
        const int pos = m % HW;
        const int y   = pos / 7, x = pos - y * 7;
        v8f acc0 = {}, acc1 = {};
#pragma unroll
        for (int tap = 0; tap < 9; ++tap) {
            const int dy = tap / 3 - 1, dx = tap % 3 - 1;
            const bool inb = (y + dy >= 0) & (y + dy < 6) & (x + dx >= 0) & (x + dx < 7);
            AFrag a;
            const v8h* p = (const v8h*)(in + (m + dy * 7 + dx) * 32 + aoff);
            a.h2[0] = inb ? p[0] : zero;
            a.h2[1] = inb ? p[1] : zero;
            acc0 = wmma_f16(a.v, Bf[tap][0], acc0);
            acc1 = wmma_f16(a.v, Bf[tap][1], acc1);
        }
        const int n  = lane & 15;
        const int mb = mt * 16 + ((lane & 16) ? 8 : 0);
        CFrag c0, c1; c0.v = acc0; c1.v = acc1;
#pragma unroll
        for (int r = 0; r < 8; ++r) {
            out[(mb + r) * 32 + n]      = (_Float16)c0.e[r];
            out[(mb + r) * 32 + n + 16] = (_Float16)c1.e[r];
        }
    }
}

// GroupNorm(16,32): thread owns (sample, channel-pair); stats stay in registers.
__device__ __forceinline__ void groupnorm(const _Float16* __restrict__ src,
                                          _Float16* __restrict__ dst,
                                          const _Float16* __restrict__ resid,
                                          const float* __restrict__ g,
                                          const float* __restrict__ b, int tid) {
    __syncthreads();
    const int s = tid >> 4, c0 = (tid & 15) * 2;
    float sum = 0.f, sq = 0.f;
    for (int p = 0; p < HW; ++p) {
        int idx = (s * HW + p) * 32 + c0;
        float v0 = (float)src[idx], v1 = (float)src[idx + 1];
        sum += v0 + v1; sq += v0 * v0 + v1 * v1;
    }
    const float m  = sum * (1.f / 84.f);
    const float rs = rsqrtf(sq * (1.f / 84.f) - m * m + 1e-5f);
    const float g0 = g[c0] * rs,     g1 = g[c0 + 1] * rs;
    const float b0 = b[c0] - m * g0, b1 = b[c0 + 1] - m * g1;
    for (int p = 0; p < HW; ++p) {
        int idx = (s * HW + p) * 32 + c0;
        float v0 = (float)src[idx]     * g0 + b0;
        float v1 = (float)src[idx + 1] * g1 + b1;
        if (resid) { v0 += (float)resid[idx]; v1 += (float)resid[idx + 1]; }
        dst[idx]     = (_Float16)fmaxf(v0, 0.f);
        dst[idx + 1] = (_Float16)fmaxf(v1, 0.f);
    }
    __syncthreads();
}

__global__ void __launch_bounds__(256)
conv_tower(const int* __restrict__ board, const _Float16* __restrict__ wsw,
           const float* __restrict__ gn0_g, const float* __restrict__ gn0_b,
           const float* __restrict__ g11, const float* __restrict__ b11,
           const float* __restrict__ g12, const float* __restrict__ b12,
           const float* __restrict__ g21, const float* __restrict__ b21,
           const float* __restrict__ g22, const float* __restrict__ b22,
           const float* __restrict__ g31, const float* __restrict__ b31,
           const float* __restrict__ g32, const float* __restrict__ b32,
           _Float16* __restrict__ f_ws) {
    extern __shared__ char smem[];
    _Float16* actA = (_Float16*)smem;   // 21504 f16
    _Float16* actB = actA + NE;         // 21504 (also conv0 im2col)
    _Float16* actC = actB + NE;         // 21504 (also board staging)

    const int tid = threadIdx.x, lane = tid & 31, wid = tid >> 5;
    const int base = blockIdx.x * TB;

    // prefetch conv weight region into cache (global_prefetch_b8)
    for (int i = tid; i < 864; i += 256)
        __builtin_prefetch((const char*)(wsw + OFF_CONV) + (size_t)i * 128, 0, 3);

    // ---- stage board tile ----
    int* bbuf = (int*)actC;
    for (int i = tid; i < NM; i += 256) bbuf[i] = board[(size_t)base * HW + i];
    __syncthreads();

    // ---- build conv0 im2col into actB: xim[m][k], k = tap*3+ci (27 real) ----
    for (int m = tid; m < NM; m += 256) {
        int pos = m % HW, y = pos / 7, x = pos - y * 7;
        X32 t;
#pragma unroll
        for (int k = 0; k < 32; ++k) {
            _Float16 v = (_Float16)0.0f;
            if (k < 27) {
                const int tap = k / 3, ci = k - tap * 3;
                const int dy = tap / 3 - 1, dx = tap % 3 - 1;
                if (y + dy >= 0 && y + dy < 6 && x + dx >= 0 && x + dx < 7) {
                    int bv = bbuf[m + dy * 7 + dx];
                    bool on = (ci == 0) ? (bv == 1) : (ci == 1) ? (bv == -1) : (bv == 0);
                    v = on ? (_Float16)1.0f : (_Float16)0.0f;
                }
            }
            t.e[k] = v;
        }
        v8h* d = (v8h*)(actB + m * 32);
#pragma unroll
        for (int j = 0; j < 4; ++j) d[j] = t.h[j];
    }
    __syncthreads();

    // ---- conv0: one k-step per M-tile; B fragments direct from global f16 ----
    {
        const int koff = (lane & 16) ? 16 : 0;
        const int aoff = (lane & 16) ? 16 : 0;
        const _Float16* w0 = wsw + OFF_W0;
        v16h B0, B1;
        {
            AFrag b;
            const v8h* p = (const v8h*)(w0 + (lane & 15) * 32 + koff);
            b.h2[0] = p[0]; b.h2[1] = p[1]; B0 = b.v;
            p = (const v8h*)(w0 + (16 + (lane & 15)) * 32 + koff);
            b.h2[0] = p[0]; b.h2[1] = p[1]; B1 = b.v;
        }
        for (int mt = wid; mt < MTILES; mt += 8) {
            const int m = mt * 16 + (lane & 15);
            AFrag a;
            const v8h* p = (const v8h*)(actB + m * 32 + aoff);
            a.h2[0] = p[0]; a.h2[1] = p[1];
            v8f acc0 = {}, acc1 = {};
            acc0 = wmma_f16(a.v, B0, acc0);
            acc1 = wmma_f16(a.v, B1, acc1);
            const int n  = lane & 15;
            const int mb = mt * 16 + ((lane & 16) ? 8 : 0);
            CFrag c0, c1; c0.v = acc0; c1.v = acc1;
#pragma unroll
            for (int r = 0; r < 8; ++r) {
                actA[(mb + r) * 32 + n]      = (_Float16)c0.e[r];
                actA[(mb + r) * 32 + n + 16] = (_Float16)c1.e[r];
            }
        }
    }
    groupnorm(actA, actA, nullptr, gn0_g, gn0_b, tid);

    const float* rg1[3]  = {g11, g21, g31};
    const float* rbb1[3] = {b11, b21, b31};
    const float* rg2[3]  = {g12, g22, g32};
    const float* rbb2[3] = {b12, b22, b32};

    for (int r = 0; r < 3; ++r) {
        conv_layer(actA, actB, wsw + OFF_CONV + (size_t)(r * 2 + 0) * 9216, lane, wid);
        groupnorm(actB, actB, nullptr, rg1[r], rbb1[r], tid);
        conv_layer(actB, actC, wsw + OFF_CONV + (size_t)(r * 2 + 1) * 9216, lane, wid);
        groupnorm(actC, actA, actA, rg2[r], rbb2[r], tid);   // h = relu(gn + h)
    }

    // ---- flatten: f[s][c*42+pos], thread owns (s, channel pair) ----
    {
        const int s = tid >> 4, c0 = (tid & 15) * 2;
        _Float16* dst = f_ws + (size_t)(base + s) * 1344;
        for (int p = 0; p < HW; ++p) {
            dst[c0 * HW + p]       = actA[(s * HW + p) * 32 + c0];
            dst[(c0 + 1) * HW + p] = actA[(s * HW + p) * 32 + c0 + 1];
        }
    }
}

// =====================  Kernel 2: dense heads (f16 weights)  =====================
template<int K, int LDA, int LDD>
__device__ __forceinline__ void gemm16(const _Float16* __restrict__ A,
                                       const _Float16* __restrict__ Wh,
                                       int ntile, float* __restrict__ dbuf, int lane) {
    v8f acc = {};
    const int srow = lane & 15;
    const int n    = ntile * 16 + (lane & 15);
    const int aoff = (lane & 16) ? 16 : 0;
    const int koff = (lane & 16) ? 16 : 0;
#pragma unroll 2
    for (int ks = 0; ks < K / 32; ++ks) {
        AFrag a, b;
        const v8h* pa = (const v8h*)(A + srow * LDA + ks * 32 + aoff);
        a.h2[0] = pa[0]; a.h2[1] = pa[1];
        const v8h* pw = (const v8h*)(Wh + (size_t)n * K + ks * 32 + koff);
        b.h2[0] = pw[0]; b.h2[1] = pw[1];
        acc = wmma_f16(a.v, b.v, acc);
    }
    CFrag c; c.v = acc;
    const int sbase = (lane & 16) ? 8 : 0;
#pragma unroll
    for (int r = 0; r < 8; ++r) dbuf[(sbase + r) * LDD + n] = c.e[r];
}

template<int NF>
__device__ __forceinline__ void ln_pass(const float* __restrict__ dbuf,
                                        const float* __restrict__ g, const float* __restrict__ b,
                                        _Float16* __restrict__ dst,
                                        float* smean, float* srstd, int tid) {
    __syncthreads();
    if (tid < 16) {
        float sum = 0.f, sq = 0.f;
        for (int k = 0; k < NF; ++k) { float v = dbuf[tid * NF + k]; sum += v; sq += v * v; }
        float m = sum * (1.f / NF);
        smean[tid] = m;
        srstd[tid] = rsqrtf(sq * (1.f / NF) - m * m + 1e-5f);
    }
    __syncthreads();
    for (int i = tid; i < 16 * NF; i += 256) {
        int s = i / NF, k = i - s * NF;
        float v = (dbuf[i] - smean[s]) * srstd[s] * g[k] + b[k];
        dst[i] = (_Float16)fmaxf(v, 0.f);
    }
    __syncthreads();
}

__global__ void __launch_bounds__(256)
dense_heads(const _Float16* __restrict__ f_ws, const _Float16* __restrict__ wsw,
            const float* __restrict__ sd_g, const float* __restrict__ sd_b,
            const float* __restrict__ p_g1, const float* __restrict__ p_b1,
            const float* __restrict__ p_g2, const float* __restrict__ p_b2,
            const float* __restrict__ p_w3, const float* __restrict__ p_bias3,
            const float* __restrict__ v_g1, const float* __restrict__ v_b1,
            const float* __restrict__ v_g2, const float* __restrict__ v_b2,
            const float* __restrict__ v_w3, const float* __restrict__ v_bias3,
            float* __restrict__ out_p, float* __restrict__ out_v) {
    __shared__ _Float16 fbuf[16 * 1344];
    __shared__ _Float16 a128[16 * 128];
    __shared__ _Float16 pbuf[16 * 128];
    __shared__ _Float16 vbuf[16 * 64];
    __shared__ float    dbuf[16 * 128];
    __shared__ float    smean[16], srstd[16];

    const int tid = threadIdx.x, lane = tid & 31, wid = tid >> 5;
    const int base = blockIdx.x * TB;

    {   // 128-bit tile copy of f
        const v8h* src = (const v8h*)(f_ws + (size_t)base * 1344);
        v8h* dst = (v8h*)fbuf;
        for (int i = tid; i < 16 * 1344 / 8; i += 256) dst[i] = src[i];
    }
    __syncthreads();

    gemm16<1344, 1344, 128>(fbuf, wsw + OFF_SD, wid, dbuf, lane);
    ln_pass<128>(dbuf, sd_g, sd_b, a128, smean, srstd, tid);

    // policy head
    gemm16<128, 128, 128>(a128, wsw + OFF_PW1, wid, dbuf, lane);
    ln_pass<128>(dbuf, p_g1, p_b1, pbuf, smean, srstd, tid);
    gemm16<128, 128, 128>(pbuf, wsw + OFF_PW2, wid, dbuf, lane);
    ln_pass<128>(dbuf, p_g2, p_b2, pbuf, smean, srstd, tid);
    if (tid < 112) {
        int s = tid / 7, j = tid - s * 7;
        float acc = p_bias3[j];
        for (int k = 0; k < 128; ++k) acc += (float)pbuf[s * 128 + k] * p_w3[j * 128 + k];
        out_p[(size_t)(base + s) * 7 + j] = acc;
    }

    // value head (4 N-tiles; wave-uniform branch keeps EXEC full inside WMMA waves)
    if (wid < 4) gemm16<128, 128, 64>(a128, wsw + OFF_VW1, wid, dbuf, lane);
    ln_pass<64>(dbuf, v_g1, v_b1, vbuf, smean, srstd, tid);
    if (wid < 4) gemm16<64, 64, 64>(vbuf, wsw + OFF_VW2, wid, dbuf, lane);
    ln_pass<64>(dbuf, v_g2, v_b2, vbuf, smean, srstd, tid);
    if (tid < 16) {
        float acc = v_bias3[0];
        for (int k = 0; k < 64; ++k) acc += (float)vbuf[tid * 64 + k] * v_w3[k];
        out_v[base + tid] = tanhf(acc);
    }
}

extern "C" void kernel_launch(void* const* d_in, const int* in_sizes, int n_in,
                              void* d_out, int out_size, void* d_ws, size_t ws_size,
                              hipStream_t stream) {
    const int*   board   = (const int*)  d_in[0];
    const float* conv0_w = (const float*)d_in[1];
    const float* gn0_g   = (const float*)d_in[2];
    const float* gn0_b   = (const float*)d_in[3];
    const float* w11=(const float*)d_in[4],  *g11=(const float*)d_in[5],  *b11=(const float*)d_in[6];
    const float* w12=(const float*)d_in[7],  *g12=(const float*)d_in[8],  *b12=(const float*)d_in[9];
    const float* w21=(const float*)d_in[10], *g21=(const float*)d_in[11], *b21=(const float*)d_in[12];
    const float* w22=(const float*)d_in[13], *g22=(const float*)d_in[14], *b22=(const float*)d_in[15];
    const float* w31=(const float*)d_in[16], *g31=(const float*)d_in[17], *b31=(const float*)d_in[18];
    const float* w32=(const float*)d_in[19], *g32=(const float*)d_in[20], *b32=(const float*)d_in[21];
    const float* sd_w=(const float*)d_in[22], *sd_g=(const float*)d_in[23], *sd_b=(const float*)d_in[24];
    const float* p_w1=(const float*)d_in[25], *p_g1=(const float*)d_in[26], *p_b1=(const float*)d_in[27];
    const float* p_w2=(const float*)d_in[28], *p_g2=(const float*)d_in[29], *p_b2=(const float*)d_in[30];
    const float* p_w3=(const float*)d_in[31], *p_bias3=(const float*)d_in[32];
    const float* v_w1=(const float*)d_in[33], *v_g1=(const float*)d_in[34], *v_b1=(const float*)d_in[35];
    const float* v_w2=(const float*)d_in[36], *v_g2=(const float*)d_in[37], *v_b2=(const float*)d_in[38];
    const float* v_w3=(const float*)d_in[39], *v_bias3=(const float*)d_in[40];

    const int B  = in_sizes[0] / (6 * 7);   // 32768
    const int nb = B / TB;                  // 2048 blocks

    _Float16* f_ws = (_Float16*)d_ws;                 // B*1344 f16 (~88 MB)
    _Float16* wsw  = f_ws + (size_t)B * 1344;         // f16 weight workspace (~0.55 MB)
    float* out_p = (float*)d_out;
    float* out_v = out_p + (size_t)B * 7;

    prep_weights<<<128, 256, 0, stream>>>(conv0_w, w11, w12, w21, w22, w31, w32,
                                          sd_w, p_w1, p_w2, v_w1, v_w2, wsw);

    const size_t smem1 = (size_t)(3 * NE) * sizeof(_Float16);   // 129 KB
    hipFuncSetAttribute((const void*)conv_tower, hipFuncAttributeMaxDynamicSharedMemorySize, (int)smem1);

    conv_tower<<<nb, 256, smem1, stream>>>(board, wsw, gn0_g, gn0_b,
        g11, b11, g12, b12, g21, b21, g22, b22, g31, b31, g32, b32, f_ws);

    dense_heads<<<nb, 256, 0, stream>>>(f_ws, wsw, sd_g, sd_b,
        p_g1, p_b1, p_g2, p_b2, p_w3, p_bias3,
        v_g1, v_b1, v_g2, v_b2, v_w3, v_bias3,
        out_p, out_v);
}
